// VGGTCrossFrameRKDDistanceLoss_36524401885589
// MI455X (gfx1250) — compile-verified, matrix operand order checked by reading, not run on hardware
//
#include <hip/hip_runtime.h>
#include <math.h>

#define B_SZ   4
#define V_T    8
#define P_TOK  1374
#define D_DIM  1024
#define NREF   256
#define EP     (4*P_TOK)                 /* 5496 teacher-only pool rows   */
#define COLT   ((EP + 15)/16)            /* 344 column tiles              */
#define CHUNKS 32                        /* column chunks (one per wave)  */
#define TPC    ((COLT + CHUNKS - 1)/CHUNKS) /* 11 tiles per chunk         */

typedef __attribute__((ext_vector_type(16))) __bf16 v16bf;
typedef __attribute__((ext_vector_type(8)))  float  v8f;

// ---------------------------------------------------------------------------
// Kernel 1: inverse L2 norms of the teacher-only candidate pool rows.
// One wave per row; B*EP = 21984 rows total.
// ---------------------------------------------------------------------------
__global__ __launch_bounds__(256)
void k_extra_norms(const float* __restrict__ teacher, float* __restrict__ invExtra) {
    const int wave = threadIdx.x >> 5, lane = threadIdx.x & 31;
    const int row  = blockIdx.x * 8 + wave;
    if (row >= B_SZ * EP) return;
    const int b = row / EP, mm = row % EP;
    const int frame = 1 + 2 * (mm / P_TOK);          // EXTRA_IDX = 1,3,5,7
    const int p     = mm % P_TOK;
    const float4* src = (const float4*)(teacher + ((size_t)(b*V_T + frame)*P_TOK + p)*(size_t)D_DIM);
    float ss = 0.0f;
    #pragma unroll
    for (int i = lane; i < D_DIM/4; i += 32) {
        float4 x = src[i];
        ss += x.x*x.x + x.y*x.y + x.z*x.z + x.w*x.w;
    }
    #pragma unroll
    for (int o = 16; o > 0; o >>= 1) ss += __shfl_xor(ss, o, 32);
    if (lane == 0) invExtra[row] = 1.0f / fmaxf(sqrtf(ss), 1e-12f);
}

// ---------------------------------------------------------------------------
// Kernel 2: cosine-sim GEMM via v_wmma_f32_16x16x32_bf16 fused with per-row
// top-4 tracking. Grid: (16 row-tiles, CHUNKS/4, B), block = 128 (4 waves,
// each wave owns an independent column chunk).
// Per-row positive scaling (ref norms) does not change top-k order, so only
// the extra-pool column norms are applied — folded into the LDS tile store
// (each lane owns one column of the C tile, so the scale is one scalar).
// ---------------------------------------------------------------------------
__global__ __launch_bounds__(128)
void k_sim_topk(const float* __restrict__ teacher, const int* __restrict__ ref_perm,
                const float* __restrict__ invExtra,
                float* __restrict__ cand_val, int* __restrict__ cand_idx) {
    __shared__ float tile[4][16][17];                 // per-wave 16x16 tile (+pad)
    const int wave = threadIdx.x >> 5, lane = threadIdx.x & 31;
    const int r    = lane & 15;                       // row-in-tile (A) / col-in-tile (B)
    const int half = lane >> 4;
    const int off  = half * 8;                        // per-lane K sub-offset
    const int rowTile = blockIdx.x;                   // 0..15
    const int b       = blockIdx.z;                   // 0..3
    const int chunk   = blockIdx.y * 4 + wave;        // 0..31

    const int refTok = ref_perm[rowTile * 16 + r];
    const float* aRow = teacher + ((size_t)(b*V_T + 0)*P_TOK + refTok)*(size_t)D_DIM;

    float tv[4] = {-1e30f, -1e30f, -1e30f, -1e30f};
    int   ti[4] = {0, 0, 0, 0};

    for (int t = 0; t < TPC; ++t) {
        const int colTile = chunk * TPC + t;
        if (colTile >= COLT) break;
        const int colBase = colTile * 16;
        int m = colBase + r; if (m > EP - 1) m = EP - 1;   // clamp pad columns
        const int frame = 1 + 2 * (m / P_TOK);
        const int p     = m % P_TOK;
        const float* bRow = teacher + ((size_t)(b*V_T + frame)*P_TOK + p)*(size_t)D_DIM;
        const float invC = invExtra[b*EP + m];        // column norm for this lane's column

        v8f acc;
        #pragma unroll
        for (int i = 0; i < 8; ++i) acc[i] = 0.0f;

        for (int kk = 0; kk < D_DIM/32; ++kk) {
            const int kb = kk*32 + off;
            const float4 ax0 = ((const float4*)(aRow + kb     ))[0];
            const float4 ax1 = ((const float4*)(aRow + kb     ))[1];
            const float4 ax2 = ((const float4*)(aRow + kb + 16))[0];
            const float4 ax3 = ((const float4*)(aRow + kb + 16))[1];
            const float4 bx0 = ((const float4*)(bRow + kb     ))[0];
            const float4 bx1 = ((const float4*)(bRow + kb     ))[1];
            const float4 bx2 = ((const float4*)(bRow + kb + 16))[0];
            const float4 bx3 = ((const float4*)(bRow + kb + 16))[1];
            v16bf A, Bm;
            A[0]=(__bf16)ax0.x;  A[1]=(__bf16)ax0.y;  A[2]=(__bf16)ax0.z;  A[3]=(__bf16)ax0.w;
            A[4]=(__bf16)ax1.x;  A[5]=(__bf16)ax1.y;  A[6]=(__bf16)ax1.z;  A[7]=(__bf16)ax1.w;
            A[8]=(__bf16)ax2.x;  A[9]=(__bf16)ax2.y;  A[10]=(__bf16)ax2.z; A[11]=(__bf16)ax2.w;
            A[12]=(__bf16)ax3.x; A[13]=(__bf16)ax3.y; A[14]=(__bf16)ax3.z; A[15]=(__bf16)ax3.w;
            Bm[0]=(__bf16)bx0.x;  Bm[1]=(__bf16)bx0.y;  Bm[2]=(__bf16)bx0.z;  Bm[3]=(__bf16)bx0.w;
            Bm[4]=(__bf16)bx1.x;  Bm[5]=(__bf16)bx1.y;  Bm[6]=(__bf16)bx1.z;  Bm[7]=(__bf16)bx1.w;
            Bm[8]=(__bf16)bx2.x;  Bm[9]=(__bf16)bx2.y;  Bm[10]=(__bf16)bx2.z; Bm[11]=(__bf16)bx2.w;
            Bm[12]=(__bf16)bx3.x; Bm[13]=(__bf16)bx3.y; Bm[14]=(__bf16)bx3.z; Bm[15]=(__bf16)bx3.w;
            acc = __builtin_amdgcn_wmma_f32_16x16x32_bf16(false, A, false, Bm,
                                                          (short)0, acc, false, false);
        }

        // C layout: acc[v] = sim[M = v + 8*half][N = r]; scale by column norm here
        #pragma unroll
        for (int v = 0; v < 8; ++v) tile[wave][v + off][r] = acc[v] * invC;
        asm volatile("s_wait_dscnt 0" ::: "memory");   // wave-local LDS ordering

        if (lane < 16) {
            const int rr = lane;
            for (int j = 0; j < 16; ++j) {
                const int col = colBase + j;
                if (col < EP) {
                    const float v = tile[wave][rr][j];
                    if (v > tv[3]) {                   // strict >: lower index wins ties
                        tv[3] = v; ti[3] = col;
                        #pragma unroll
                        for (int s = 3; s > 0; --s)
                            if (tv[s] > tv[s-1]) {
                                float fv = tv[s]; tv[s] = tv[s-1]; tv[s-1] = fv;
                                int   iv = ti[s]; ti[s] = ti[s-1]; ti[s-1] = iv;
                            }
                    }
                }
            }
        }
        asm volatile("s_wait_dscnt 0" ::: "memory");   // reads done before next tile store
    }

    if (lane < 16) {
        const size_t base = (((size_t)b*NREF + rowTile*16 + lane)*CHUNKS + chunk)*4;
        #pragma unroll
        for (int s = 0; s < 4; ++s) { cand_val[base+s] = tv[s]; cand_idx[base+s] = ti[s]; }
    }
}

// ---------------------------------------------------------------------------
// Kernel 3: merge per-chunk top-4 lists into final top-4 indices per (b,n).
// Chunks partition columns, so candidates are duplicate-free.
// ---------------------------------------------------------------------------
__global__ __launch_bounds__(256)
void k_merge_topk(const float* __restrict__ cand_val, const int* __restrict__ cand_idx,
                  int* __restrict__ topk) {
    const int tid = blockIdx.x * 256 + threadIdx.x;    // 0..B*NREF-1
    if (tid >= B_SZ * NREF) return;
    const size_t base = (size_t)tid * CHUNKS * 4;
    int c0 = -1, c1 = -1, c2 = -1;
    for (int s = 0; s < 4; ++s) {
        float best = -3e38f; int bestSlot = -1; int bestIdx = 0x7fffffff;
        for (int e = 0; e < CHUNKS*4; ++e) {
            if (e == c0 || e == c1 || e == c2) continue;
            const float v = cand_val[base + e];
            const int  ix = cand_idx[base + e];
            if (v > best || (v == best && ix < bestIdx)) { best = v; bestSlot = e; bestIdx = ix; }
        }
        topk[tid*4 + s] = bestIdx;
        if      (s == 0) c0 = bestSlot;
        else if (s == 1) c1 = bestSlot;
        else if (s == 2) c2 = bestSlot;
    }
}

// ---------------------------------------------------------------------------
// Block reduction helpers (256 threads = 8 waves of 32).
// ---------------------------------------------------------------------------
__device__ __forceinline__ float blk_max(float v, float* buf) {
    #pragma unroll
    for (int o = 16; o > 0; o >>= 1) v = fmaxf(v, __shfl_xor(v, o, 32));
    __syncthreads();
    if ((threadIdx.x & 31) == 0) buf[threadIdx.x >> 5] = v;
    __syncthreads();
    float r = buf[0];
    #pragma unroll
    for (int w = 1; w < 8; ++w) r = fmaxf(r, buf[w]);
    return r;
}
__device__ __forceinline__ float blk_sum(float v, float* buf) {
    #pragma unroll
    for (int o = 16; o > 0; o >>= 1) v += __shfl_xor(v, o, 32);
    __syncthreads();
    if ((threadIdx.x & 31) == 0) buf[threadIdx.x >> 5] = v;
    __syncthreads();
    float r = buf[0];
    #pragma unroll
    for (int w = 1; w < 8; ++w) r += buf[w];
    return r;
}

// KL(softmax(ta-tb) || softmax(sa-sb)) over D channels, TEMP=1.
// KL = A/Zt - (mt + log Zt) + (ms + log Zs),  A = sum e^(dt-mt)*(dt-ds)
__device__ __forceinline__ float kl_block(const float* ta, const float* tb,
                                          const float* sa, const float* sb, float* buf) {
    const int tid = threadIdx.x;
    float mt = -3e38f, ms = -3e38f;
    for (int i = tid; i < D_DIM; i += 256) {
        const float dt = ta[i] - tb[i], ds = sa[i] - sb[i];
        mt = fmaxf(mt, dt); ms = fmaxf(ms, ds);
    }
    mt = blk_max(mt, buf); ms = blk_max(ms, buf);
    float zt = 0.0f, zs = 0.0f, ac = 0.0f;
    for (int i = tid; i < D_DIM; i += 256) {
        const float dt = ta[i] - tb[i], ds = sa[i] - sb[i];
        const float e = __expf(dt - mt);
        zt += e; ac += e * (dt - ds); zs += __expf(ds - ms);
    }
    zt = blk_sum(zt, buf); zs = blk_sum(zs, buf); ac = blk_sum(ac, buf);
    return ac / zt - (mt + __logf(zt)) + (ms + __logf(zs));
}

__device__ __forceinline__ float huber05(float x) {   // smooth_l1, beta = 0.5
    const float a = fabsf(x);
    return (a < 0.5f) ? (x * x) : (a - 0.25f);
}

// ---------------------------------------------------------------------------
// Kernel 4: one block per (b,n). Stage 11 rows in LDS, compute 19 KL terms.
// d2 is loop-invariant in the reference's shared-frame loop: computed once.
// ---------------------------------------------------------------------------
__global__ __launch_bounds__(256)
void k_loss(const float* __restrict__ teacher, const float* __restrict__ student,
            const int* __restrict__ ref_perm, const int* __restrict__ shared_perm,
            const int* __restrict__ topk, float* __restrict__ partial) {
    __shared__ float sm[11 * D_DIM];
    __shared__ float buf[8];
    const int n = blockIdx.x & 255;
    const int b = blockIdx.x >> 8;
    const int tid = threadIdx.x;
    float* refT = sm;
    float* refS = sm + D_DIM;
    float* stj  = sm + 2*D_DIM;   // 3 teacher shared rows
    float* ssj  = sm + 5*D_DIM;   // 3 student shared rows
    float* nbr  = sm + 8*D_DIM;   // 4 neighbor rows

    const int rtok = ref_perm[n];
    const int stok = shared_perm[n];
    const float* pT0 = teacher + ((size_t)(b*V_T + 0)*P_TOK + rtok)*(size_t)D_DIM;
    const float* pS0 = student + ((size_t)(b*4   + 0)*P_TOK + rtok)*(size_t)D_DIM;
    for (int i = tid; i < D_DIM; i += 256) { refT[i] = pT0[i]; refS[i] = pS0[i]; }
    for (int j = 0; j < 3; ++j) {                       // SHARED_T=2,4,6 ; SHARED_S=1,2,3
        const float* pt = teacher + ((size_t)(b*V_T + (2 + 2*j))*P_TOK + stok)*(size_t)D_DIM;
        const float* ps = student + ((size_t)(b*4   + (1 +   j))*P_TOK + stok)*(size_t)D_DIM;
        for (int i = tid; i < D_DIM; i += 256) { stj[j*D_DIM+i] = pt[i]; ssj[j*D_DIM+i] = ps[i]; }
    }
    for (int k = 0; k < 4; ++k) {
        const int m = topk[((size_t)b*NREF + n)*4 + k];
        const int frame = 1 + 2*(m / P_TOK), p = m % P_TOK;
        const float* pn = teacher + ((size_t)(b*V_T + frame)*P_TOK + p)*(size_t)D_DIM;
        for (int i = tid; i < D_DIM; i += 256) nbr[k*D_DIM+i] = pn[i];
    }
    __syncthreads();

    float s1 = 0.0f, s2 = 0.0f, s3 = 0.0f;
    for (int j = 0; j < 3; ++j) {
        const float kl = kl_block(refT, stj + j*D_DIM, refS, ssj + j*D_DIM, buf);
        if (tid == 0) s1 += huber05(kl);
    }
    for (int k = 0; k < 4; ++k) {
        const float kl = kl_block(refT, nbr + k*D_DIM, refS, nbr + k*D_DIM, buf);
        if (tid == 0) s2 += huber05(kl);
    }
    for (int j = 0; j < 3; ++j)
        for (int k = 0; k < 4; ++k) {
            const float kl = kl_block(stj + j*D_DIM, nbr + k*D_DIM, ssj + j*D_DIM, nbr + k*D_DIM, buf);
            if (tid == 0) s3 += huber05(kl);
        }
    if (tid == 0) {
        partial[blockIdx.x*3 + 0] = s1;
        partial[blockIdx.x*3 + 1] = s2;
        partial[blockIdx.x*3 + 2] = s3;
    }
}

// ---------------------------------------------------------------------------
// Kernel 5: deterministic final reduction.
// loss = S1/(3*B*N) + (3*S2)/(3*B*N*k) + S3/(3*B*N*k)
//      = S1/3072 + S2/4096 + S3/12288
// ---------------------------------------------------------------------------
__global__ __launch_bounds__(256)
void k_finalize(const float* __restrict__ partial, float* __restrict__ out) {
    __shared__ float buf[8];
    float s1 = 0.0f, s2 = 0.0f, s3 = 0.0f;
    for (int i = threadIdx.x; i < B_SZ*NREF; i += 256) {
        s1 += partial[i*3 + 0]; s2 += partial[i*3 + 1]; s3 += partial[i*3 + 2];
    }
    s1 = blk_sum(s1, buf); s2 = blk_sum(s2, buf); s3 = blk_sum(s3, buf);
    if (threadIdx.x == 0)
        out[0] = s1 * (1.0f/3072.0f) + s2 * (1.0f/4096.0f) + s3 * (1.0f/12288.0f);
}

// ---------------------------------------------------------------------------
extern "C" void kernel_launch(void* const* d_in, const int* in_sizes, int n_in,
                              void* d_out, int out_size, void* d_ws, size_t ws_size,
                              hipStream_t stream) {
    (void)in_sizes; (void)n_in; (void)out_size; (void)ws_size;
    const float* teacher     = (const float*)d_in[0];
    const float* student     = (const float*)d_in[1];
    const int*   ref_perm    = (const int*)d_in[2];
    const int*   shared_perm = (const int*)d_in[3];
    float* out = (float*)d_out;

    // workspace layout (~1.15 MB)
    char* ws = (char*)d_ws;
    size_t off = 0;
    float* invExtra = (float*)(ws + off); off += sizeof(float) * B_SZ * EP;
    off = (off + 255) & ~(size_t)255;
    float* cand_val = (float*)(ws + off); off += sizeof(float) * B_SZ * NREF * CHUNKS * 4;
    int*   cand_idx = (int*)  (ws + off); off += sizeof(int)   * B_SZ * NREF * CHUNKS * 4;
    int*   topk     = (int*)  (ws + off); off += sizeof(int)   * B_SZ * NREF * 4;
    float* partial  = (float*)(ws + off);

    hipLaunchKernelGGL(k_extra_norms, dim3((B_SZ*EP)/8), dim3(256), 0, stream,
                       teacher, invExtra);
    hipLaunchKernelGGL(k_sim_topk, dim3(16, CHUNKS/4, B_SZ), dim3(128), 0, stream,
                       teacher, ref_perm, invExtra, cand_val, cand_idx);
    hipLaunchKernelGGL(k_merge_topk, dim3((B_SZ*NREF + 255)/256), dim3(256), 0, stream,
                       cand_val, cand_idx, topk);
    hipLaunchKernelGGL(k_loss, dim3(B_SZ*NREF), dim3(256), 0, stream,
                       teacher, student, ref_perm, shared_perm, topk, partial);
    hipLaunchKernelGGL(k_finalize, dim3(1), dim3(256), 0, stream, partial, out);
}
